// BertImageSelfAttention_16277926052412
// MI455X (gfx1250) — matrix operand order repaired and checked by reading, not compiled, not run on hardware
//
#include <hip/hip_runtime.h>
#include <hip/hip_bf16.h>
#include <math.h>

// ---------------------------------------------------------------------------
// Problem constants: B=4, S=2048, DV=1024, H=16, Dh=64, T=64, DT=768.
// ---------------------------------------------------------------------------
#define BB   4
#define SS   2048
#define DVV  1024
#define HH   16
#define DHH  64
#define TT   64
#define DTT  768

typedef __attribute__((ext_vector_type(16))) _Float16 v16h;
typedef __attribute__((ext_vector_type(8)))  _Float16 v8h;
typedef __attribute__((ext_vector_type(8)))  float    v8f;

union FragA { v16h v; v8h h[2]; };

// ---------------------------------------------------------------------------
// Kernel 1: masked-mean pool of text embedding -> pool [B, DT]
// ---------------------------------------------------------------------------
__global__ __launch_bounds__(256)
void pool_kernel(const float* __restrict__ txt, const float* __restrict__ tmask,
                 float* __restrict__ pool) {
    int b = blockIdx.x;
    float msum = 0.0f;
    for (int t = 0; t < TT; ++t) msum += tmask[b * TT + t];
    float inv = 1.0f / msum;
    for (int d = threadIdx.x; d < DTT; d += blockDim.x) {
        float s = 0.0f;
        for (int t = 0; t < TT; ++t)
            s += txt[((size_t)b * TT + t) * DTT + d] * tmask[b * TT + t];
        pool[b * DTT + d] = s * inv;
    }
}

// ---------------------------------------------------------------------------
// Kernel 2: gate = 1 + sigmoid(pool @ Wd + bd)   -> gate [B, DV]
// ---------------------------------------------------------------------------
__global__ __launch_bounds__(256)
void gate_kernel(const float* __restrict__ pool, const float* __restrict__ Wd,
                 const float* __restrict__ bd, float* __restrict__ gate) {
    int b = blockIdx.x;
    int o = blockIdx.y * 256 + threadIdx.x;
    float s = bd[o];
    const float* p = pool + b * DTT;
    for (int i = 0; i < DTT; ++i)
        s += p[i] * Wd[(size_t)i * DVV + o];
    gate[b * DVV + o] = 1.0f + 1.0f / (1.0f + __expf(-s));
}

// ---------------------------------------------------------------------------
// Prep kernels: one-shot f32 -> f16 conversion so the GEMM can stage raw
// f16 bytes with async global->LDS copies (no VGPR round trip).
// ---------------------------------------------------------------------------
__global__ __launch_bounds__(256)
void convert_f16(const float* __restrict__ src, _Float16* __restrict__ dst) {
    size_t i = ((size_t)blockIdx.x * 256 + threadIdx.x) * 4;
    float4 f = *(const float4*)(src + i);
    dst[i + 0] = (_Float16)f.x; dst[i + 1] = (_Float16)f.y;
    dst[i + 2] = (_Float16)f.z; dst[i + 3] = (_Float16)f.w;
}

// W [k][n] f32 -> Wt [n][k] f16  (64x64 tiles via LDS)
__global__ __launch_bounds__(256)
void transpose_f16(const float* __restrict__ W, _Float16* __restrict__ Wt) {
    __shared__ _Float16 t[64][72];
    int k0 = blockIdx.y * 64, n0 = blockIdx.x * 64;
    int tid = threadIdx.x;
    int r  = tid >> 2;           // 0..63
    int c0 = (tid & 3) * 16;     // 0,16,32,48
#pragma unroll
    for (int j = 0; j < 16; j += 4) {
        float4 f = *(const float4*)(W + (size_t)(k0 + r) * DVV + n0 + c0 + j);
        t[c0 + j + 0][r] = (_Float16)f.x;
        t[c0 + j + 1][r] = (_Float16)f.y;
        t[c0 + j + 2][r] = (_Float16)f.z;
        t[c0 + j + 3][r] = (_Float16)f.w;
    }
    __syncthreads();
#pragma unroll
    for (int j = 0; j < 16; ++j)
        Wt[(size_t)(n0 + r) * DVV + k0 + c0 + j] = t[r][c0 + j];
}

// ---------------------------------------------------------------------------
// Async global->LDS staging helpers (CDNA5 ASYNCcnt path, ISA ch.10/§15.18).
// The instruction's IOFFSET applies to BOTH the LDS and global address, so a
// 32B row chunk is two b128 asyncs off one base.  Generic __shared__ pointers
// carry the LDS byte offset in their low 32 bits (LDS aperture).
// ---------------------------------------------------------------------------
__device__ __forceinline__ void async_copy_32B(const _Float16* gsrc, _Float16* ldst) {
    unsigned l = (unsigned)(uintptr_t)ldst;
    asm volatile("global_load_async_to_lds_b128 %0, %1, off"
                 :: "v"(l), "v"(gsrc) : "memory");
    asm volatile("global_load_async_to_lds_b128 %0, %1, off offset:16"
                 :: "v"(l), "v"(gsrc) : "memory");
}
__device__ __forceinline__ void wait_async0() {
    asm volatile("s_wait_asynccnt 0x0" ::: "memory");
}

// ---------------------------------------------------------------------------
// Kernel 3: WMMA projection GEMM, D = A[8192,1024] @ W[1024,1024] + bias.
// A f16 row-major; Wt f16 pre-transposed [n][k].  Block: 256 threads
// (8 waves), tile 128(M) x 128(N), K-step 32, double-buffered LDS filled by
// GLOBAL_LOAD_ASYNC_TO_LDS_B128.  Wave w: rows (w/2)*32, cols (w%2)*64 ->
// 2x4 WMMA tiles, 8 WMMAs per K-step.
// MODE 0/1: gated, out f16 [B,H,S,Dh].  MODE 2: ungated, out f16 [B,H,Dh,S].
// ---------------------------------------------------------------------------
template <int MODE>
__global__ __launch_bounds__(256)
void qkv_gemm(const _Float16* __restrict__ A, const _Float16* __restrict__ Wt,
              const float* __restrict__ bias, const float* __restrict__ gate,
              _Float16* __restrict__ out) {
    __shared__ _Float16 Atile[2][128][32];   // [m][k], rows = 64B
    __shared__ _Float16 Btile[2][128][32];   // [n][k], rows = 64B

    const int tid  = threadIdx.x;
    const int lane = tid & 31;
    const int wave = tid >> 5;
    const int wm   = wave >> 1;        // 0..3  (row group)
    const int wn   = wave & 1;         // 0..1  (col group)
    const int hi   = lane >> 4;
    const int lm   = lane & 15;

    const int rowBase = blockIdx.y * 128;
    const int colBase = blockIdx.x * 128;

    // staging assignment: 2 threads per row, 32B (16 halfs) each
    const int sr  = tid >> 1;          // 0..127
    const int soff = (tid & 1) * 16;   // halfs

    v8f acc[2][4] = {};

    // prologue: stage tile 0
    async_copy_32B(A  + (size_t)(rowBase + sr) * DVV + soff, &Atile[0][sr][soff]);
    async_copy_32B(Wt + (size_t)(colBase + sr) * DVV + soff, &Btile[0][sr][soff]);

    for (int kt = 0; kt < DVV / 32; ++kt) {
        wait_async0();          // this wave's async copies landed
        __syncthreads();        // everyone's copies landed

        if (kt + 1 < DVV / 32) {
            int nb = (kt + 1) & 1, nk = (kt + 1) * 32;
            async_copy_32B(A  + (size_t)(rowBase + sr) * DVV + nk + soff, &Atile[nb][sr][soff]);
            async_copy_32B(Wt + (size_t)(colBase + sr) * DVV + nk + soff, &Btile[nb][sr][soff]);
        }

        const int buf = kt & 1;
        FragA fa[2], fb[4];
#pragma unroll
        for (int i = 0; i < 2; ++i) {
            int m = wm * 32 + i * 16 + lm;
            fa[i].h[0] = *(const v8h*)&Atile[buf][m][hi * 16];      // K = 16*hi+0..7
            fa[i].h[1] = *(const v8h*)&Atile[buf][m][hi * 16 + 8];  // K = 16*hi+8..15
        }
#pragma unroll
        for (int j = 0; j < 4; ++j) {
            int n = wn * 64 + j * 16 + lm;
            fb[j].h[0] = *(const v8h*)&Btile[buf][n][hi * 16];      // K = 16*hi+0..7
            fb[j].h[1] = *(const v8h*)&Btile[buf][n][hi * 16 + 8];  // K = 16*hi+8..15
        }
#pragma unroll
        for (int i = 0; i < 2; ++i)
#pragma unroll
            for (int j = 0; j < 4; ++j)
                acc[i][j] = __builtin_amdgcn_wmma_f32_16x16x32_f16(
                    false, fa[i].v, false, fb[j].v, (short)0, acc[i][j], false, false);
        __syncthreads();        // reads done before this buffer is refilled
    }

    // --- epilogue (compile-time MODE, branch-free) ---
#pragma unroll
    for (int i = 0; i < 2; ++i)
#pragma unroll
        for (int j = 0; j < 4; ++j) {
            int gc = colBase + wn * 64 + j * 16 + lm;
            float bv = bias[gc];
            int h = gc >> 6, dh = gc & 63;
#pragma unroll
            for (int r = 0; r < 8; ++r) {
                int gr = rowBase + wm * 32 + i * 16 + hi * 8 + r;
                int b = gr >> 11, s = gr & (SS - 1);
                float val = acc[i][j][r] + bv;
                if (MODE < 2) {
                    val *= gate[b * DVV + gc];
                    out[(((size_t)(b * HH + h) * SS + s) * DHH) + dh] = (_Float16)val;
                } else {
                    out[(((size_t)(b * HH + h) * DHH + dh) * SS) + s] = (_Float16)val;
                }
            }
        }
}

// ---------------------------------------------------------------------------
// Kernel 4: flash attention per (b,h).  One wave owns 32 query rows.
// Per 32-key chunk: 8 score WMMAs, online softmax with all 32 lanes (one row
// each), 8 context WMMAs; K/V fragments amortized over two 16-row tiles.
// ---------------------------------------------------------------------------
__global__ __launch_bounds__(32)
void attn_kernel(const _Float16* __restrict__ Q, const _Float16* __restrict__ K,
                 const _Float16* __restrict__ Vt, const float* __restrict__ amask,
                 float* __restrict__ out) {
    const int bh   = blockIdx.x;          // 0..B*H-1
    const int b    = bh >> 4;
    const int qt   = blockIdx.y;          // 32-query tile, 0..S/32-1
    const int lane = threadIdx.x;
    const int hi   = lane >> 4;
    const int lm   = lane & 15;

    __shared__ float    sb[32][33];       // raw scores (padded vs bank conflicts)
    __shared__ _Float16 pb[32][40];       // exp(scores), 16B-aligned 8-half runs

    // Q fragments: row subtiles i=0,1 ; dh chunks f=0,1 (A layout)
    FragA q[2][2];
#pragma unroll
    for (int i = 0; i < 2; ++i) {
        const _Float16* Qb = Q + ((size_t)bh * SS + qt * 32 + i * 16 + lm) * DHH;
        q[i][0].h[0] = *(const v8h*)(Qb + hi * 16);
        q[i][0].h[1] = *(const v8h*)(Qb + hi * 16 + 8);
        q[i][1].h[0] = *(const v8h*)(Qb + 32 + hi * 16);
        q[i][1].h[1] = *(const v8h*)(Qb + 32 + hi * 16 + 8);
    }

    v8f ctx[2][4] = {};
    float m_i = -1e30f, l_i = 0.0f;       // every lane owns row = lane
    const float scale = 0.125f;           // 1/sqrt(64)

    for (int j0 = 0; j0 < SS; j0 += 32) {
        // prefetch next chunk's K/V into cache (global_prefetch_b8)
        if (j0 + 32 < SS) {
            __builtin_prefetch(K + ((size_t)bh * SS + j0 + 32 + lm) * DHH, 0, 0);
            __builtin_prefetch(Vt + ((size_t)bh * DHH + lm) * SS + j0 + 32, 0, 0);
        }
        // ---- scores: 2 key subtiles x 2 row subtiles ----
#pragma unroll
        for (int t = 0; t < 2; ++t) {
            int key0 = j0 + t * 16;
            const _Float16* Kb = K + ((size_t)bh * SS + key0 + lm) * DHH;
            FragA kf[2];                                 // B layout: n=key, K=dh
            kf[0].h[0] = *(const v8h*)(Kb + hi * 16);
            kf[0].h[1] = *(const v8h*)(Kb + hi * 16 + 8);
            kf[1].h[0] = *(const v8h*)(Kb + 32 + hi * 16);
            kf[1].h[1] = *(const v8h*)(Kb + 32 + hi * 16 + 8);
            float mv = amask[(size_t)b * SS + key0 + lm];
#pragma unroll
            for (int i = 0; i < 2; ++i) {
                v8f s = {};
                s = __builtin_amdgcn_wmma_f32_16x16x32_f16(false, q[i][0].v, false, kf[0].v,
                                                           (short)0, s, false, false);
                s = __builtin_amdgcn_wmma_f32_16x16x32_f16(false, q[i][1].v, false, kf[1].v,
                                                           (short)0, s, false, false);
#pragma unroll
                for (int r = 0; r < 8; ++r)
                    sb[i * 16 + hi * 8 + r][t * 16 + lm] = s[r] * scale + mv;
            }
        }
        __syncthreads();

        // ---- online softmax: all 32 lanes, row = lane ----
        float mx = m_i;
#pragma unroll
        for (int c = 0; c < 32; ++c) mx = fmaxf(mx, sb[lane][c]);
        float corr = __expf(m_i - mx);
        float lsum = 0.0f;
#pragma unroll
        for (int c = 0; c < 32; ++c) {
            float e = __expf(sb[lane][c] - mx);
            pb[lane][c] = (_Float16)e;
            lsum += e;
        }
        l_i = l_i * corr + lsum;
        m_i = mx;
        __syncthreads();

        // ---- rescale accumulators by per-row correction ----
#pragma unroll
        for (int i = 0; i < 2; ++i)
#pragma unroll
            for (int r = 0; r < 8; ++r) {
                float cr = __shfl(corr, i * 16 + hi * 8 + r, 32);
                ctx[i][0][r] *= cr; ctx[i][1][r] *= cr;
                ctx[i][2][r] *= cr; ctx[i][3][r] *= cr;
            }

        // ---- P fragments (A layout) from LDS ----
        FragA p[2];
#pragma unroll
        for (int i = 0; i < 2; ++i) {
            p[i].h[0] = *(const v8h*)&pb[i * 16 + lm][hi * 16];
            p[i].h[1] = *(const v8h*)&pb[i * 16 + lm][hi * 16 + 8];
        }

        // ---- context: 4 dh tiles of V^T, shared by both row subtiles ----
#pragma unroll
        for (int n = 0; n < 4; ++n) {
            const _Float16* Vb = Vt + ((size_t)bh * DHH + n * 16 + lm) * SS + j0 + hi * 16;
            FragA vf;
            vf.h[0] = *(const v8h*)(Vb);
            vf.h[1] = *(const v8h*)(Vb + 8);
#pragma unroll
            for (int i = 0; i < 2; ++i)
                ctx[i][n] = __builtin_amdgcn_wmma_f32_16x16x32_f16(
                    false, p[i].v, false, vf.v, (short)0, ctx[i][n], false, false);
        }
        __syncthreads();
    }

    // ---- finalize: divide by row sum, write f32 output [B,S,DV] ----
    float linv = 1.0f / l_i;
    const int h = bh & 15;
#pragma unroll
    for (int i = 0; i < 2; ++i)
#pragma unroll
        for (int r = 0; r < 8; ++r) {
            float li = __shfl(linv, i * 16 + hi * 8 + r, 32);
            int s = qt * 32 + i * 16 + hi * 8 + r;
            float* orow = out + ((size_t)b * SS + s) * DVV + h * DHH;
#pragma unroll
            for (int n = 0; n < 4; ++n)
                orow[n * 16 + lm] = ctx[i][n][r] * li;
        }
}

// ---------------------------------------------------------------------------
// Host-side launcher
// ---------------------------------------------------------------------------
extern "C" void kernel_launch(void* const* d_in, const int* in_sizes, int n_in,
                              void* d_out, int out_size, void* d_ws, size_t ws_size,
                              hipStream_t stream) {
    (void)in_sizes; (void)n_in; (void)out_size; (void)ws_size;

    const float* hs    = (const float*)d_in[0];   // [B,S,DV]
    const float* amask = (const float*)d_in[1];   // [B,1,1,S]
    const float* txt   = (const float*)d_in[2];   // [B,T,DT]
    const float* tmask = (const float*)d_in[3];   // [B,T,1]
    const float* Wq    = (const float*)d_in[4];
    const float* bq    = (const float*)d_in[5];
    const float* Wk    = (const float*)d_in[6];
    const float* bk    = (const float*)d_in[7];
    const float* Wv    = (const float*)d_in[8];
    const float* bv    = (const float*)d_in[9];
    const float* Wdq   = (const float*)d_in[10];
    const float* bdq   = (const float*)d_in[11];
    const float* Wdk   = (const float*)d_in[12];
    const float* bdk   = (const float*)d_in[13];
    float* out = (float*)d_out;

    // workspace carve-up
    const size_t NHALF = (size_t)BB * SS * DVV;   // 8M halfs
    const size_t WHALF = (size_t)DVV * DVV;       // 1M halfs
    _Float16* qbuf  = (_Float16*)d_ws;            // 16 MB
    _Float16* kbuf  = qbuf + NHALF;               // 16 MB
    _Float16* vtbuf = kbuf + NHALF;               // 16 MB
    _Float16* ah    = vtbuf + NHALF;              // 16 MB  (hidden f16)
    _Float16* wtq   = ah + NHALF;                 // 2 MB   (Wq^T f16)
    _Float16* wtk   = wtq + WHALF;                // 2 MB
    _Float16* wtv   = wtk + WHALF;                // 2 MB
    float* pool = (float*)(wtv + WHALF);
    float* gq   = pool + BB * DTT;
    float* gk   = gq + BB * DVV;

    // 1) pooled text embedding + dynamic gates
    pool_kernel<<<BB, 256, 0, stream>>>(txt, tmask, pool);
    gate_kernel<<<dim3(BB, DVV / 256), 256, 0, stream>>>(pool, Wdq, bdq, gq);
    gate_kernel<<<dim3(BB, DVV / 256), 256, 0, stream>>>(pool, Wdk, bdk, gk);

    // 2) one-shot precision prep: A -> f16, W -> f16 transposed [n][k]
    convert_f16<<<NHALF / 1024, 256, 0, stream>>>(hs, ah);
    dim3 tgrid(DVV / 64, DVV / 64);
    transpose_f16<<<tgrid, 256, 0, stream>>>(Wq, wtq);
    transpose_f16<<<tgrid, 256, 0, stream>>>(Wk, wtk);
    transpose_f16<<<tgrid, 256, 0, stream>>>(Wv, wtv);

    // 3) Q/K/V projections (WMMA + async LDS staging). grid = (N/128, M/128)
    dim3 ggrid(DVV / 128, (BB * SS) / 128);
    qkv_gemm<0><<<ggrid, 256, 0, stream>>>(ah, wtq, bq, gq, qbuf);
    qkv_gemm<1><<<ggrid, 256, 0, stream>>>(ah, wtk, bk, gk, kbuf);
    qkv_gemm<2><<<ggrid, 256, 0, stream>>>(ah, wtv, bv, nullptr, vtbuf);

    // 4) flash attention (WMMA), one wave per 32 query rows
    attn_kernel<<<dim3(BB * HH, SS / 32), 32, 0, stream>>>(qbuf, kbuf, vtbuf, amask, out);
}
